// NodeEncoder_90056874262535
// MI455X (gfx1250) — matrix operand order repaired and checked by reading, not compiled, not run on hardware
//
#include <hip/hip_runtime.h>
#include <hip/hip_bf16.h>

// Problem sizes (fixed by the reference).
#define N_NODES 50000
#define N_EDGES 800000
#define UDIM    64
#define DEPTH   4
#define WV      8               // waves per 256-thread block (wave32)
#define NTILES_N (N_NODES/16)   // 3125
#define NTILES_E (N_EDGES/16)   // 50000

typedef __attribute__((ext_vector_type(16))) __bf16 v16bf;
typedef __attribute__((ext_vector_type(8)))  float  v8f;

union BF16x16 { v16bf v; uint4 u[2]; __bf16 e[16]; };

__device__ __forceinline__ unsigned short f2bf_bits(float f) {
  union { __bf16 b; unsigned short s; } u; u.b = (__bf16)f; return u.s;
}
__device__ __forceinline__ float silu_f(float x) { return x / (1.0f + __expf(-x)); }

// A-operand slice loader (16x32 bf16, MxK): lane half hf holds K = {K0+hf*8..+7, K0+16+hf*8..+7}.
// Both runs are contiguous -> two b128 loads (global or ds, inferred from pointer).
__device__ __forceinline__ v16bf ldA_bf(const unsigned short* row, int K0, int hf) {
  BF16x16 r;
  r.u[0] = *(const uint4*)(row + K0 + hf * 8);
  r.u[1] = *(const uint4*)(row + K0 + 16 + hf * 8);
  return r.v;
}
// Same A slice but sourced from f32 (aggr), scaled by invdeg, converted to bf16.
__device__ __forceinline__ v16bf ldA_f32scaled(const float* row, int K0, int hf, float s) {
  BF16x16 r;
  const float* p0 = row + K0 + hf * 8;
  const float* p1 = row + K0 + 16 + hf * 8;
#pragma unroll
  for (int i = 0; i < 8; i++) {
    r.e[i]     = (__bf16)(p0[i] * s);
    r.e[8 + i] = (__bf16)(p1[i] * s);
  }
  return r.v;
}
// B-operand loader from packed weights (LDS-resident): 32 contiguous bytes per lane.
__device__ __forceinline__ v16bf ldB_packed(const unsigned short* packed, int kslice, int t, int lane) {
  const unsigned short* p = packed + (((kslice * 4 + t) * 32) + lane) * 16;
  BF16x16 r;
  r.u[0] = ((const uint4*)p)[0];
  r.u[1] = ((const uint4*)p)[1];
  return r.v;
}
__device__ __forceinline__ v8f wmma_bf16(v16bf a, v16bf b, v8f c) {
  return __builtin_amdgcn_wmma_f32_16x16x32_bf16(false, a, false, b, (short)0, c, false, false);
}

// ---------------- small prep kernels ----------------

// Pack a (K x 64) f32 weight into bf16 B-operand layout:
// packed[((kslice*4+t)*32 + lane)*16 + i] = W[(kslice*32 + (i>>3)*16 + (lane>>4)*8 + (i&7))*64 + t*16 + (lane&15)]
__global__ void pack_w_kernel(const float* __restrict__ W, unsigned short* __restrict__ out, int K) {
  int tid = blockIdx.x * blockDim.x + threadIdx.x;
  if (tid >= K * 64) return;
  int i = tid & 15;
  int lane = (tid >> 4) & 31;
  int t = (tid >> 9) & 3;
  int kslice = tid >> 11;
  int k = kslice * 32 + (i >> 3) * 16 + (lane >> 4) * 8 + (i & 7);
  int n = t * 16 + (lane & 15);
  out[tid] = f2bf_bits(W[k * 64 + n]);
}

__global__ void zero_kernel(float* __restrict__ aggr, float* __restrict__ deg) {
  size_t i = (size_t)blockIdx.x * blockDim.x + threadIdx.x;
  if (i < (size_t)N_NODES * UDIM) aggr[i] = 0.0f;
  else if (i < (size_t)N_NODES * UDIM + N_NODES) deg[i - (size_t)N_NODES * UDIM] = 0.0f;
}

__global__ void node_enc_kernel(const float* __restrict__ x, const float* __restrict__ W,
                                const float* __restrict__ b, float* __restrict__ hf,
                                unsigned short* __restrict__ hbf) {
  int n = blockIdx.x * blockDim.x + threadIdx.x;
  if (n >= N_NODES) return;
  float x0 = x[n * 2], x1 = x[n * 2 + 1];
#pragma unroll 8
  for (int u = 0; u < UDIM; u++) {
    float v = silu_f(x0 * W[u] + x1 * W[UDIM + u] + b[u]);
    hf[(size_t)n * UDIM + u] = v;
    hbf[(size_t)n * UDIM + u] = f2bf_bits(v);
  }
}

__global__ void edge_enc_kernel(const float* __restrict__ attr, const int* __restrict__ ei,
                                const float* __restrict__ W, const float* __restrict__ b,
                                unsigned short* __restrict__ ebf, float* __restrict__ deg) {
  int ed = blockIdx.x * blockDim.x + threadIdx.x;
  if (ed >= N_EDGES) return;
  float a = attr[ed];
#pragma unroll 8
  for (int u = 0; u < UDIM; u++)
    ebf[(size_t)ed * UDIM + u] = f2bf_bits(silu_f(a * W[u] + b[u]));
  atomicAdd(&deg[ei[N_EDGES + ed]], 1.0f);
}

__global__ void invdeg_kernel(const float* __restrict__ deg, float* __restrict__ invd) {
  int n = blockIdx.x * blockDim.x + threadIdx.x;
  if (n < N_NODES) invd[n] = 1.0f / fmaxf(deg[n], 1.0f);
}

// ---------------- message kernel: one wave per TWO 16-edge tiles ----------------
// Weights live in LDS (loaded once per block); each B ds_load feeds 2 WMMAs.
__global__ __launch_bounds__(256) void msg_kernel(
    const unsigned short* __restrict__ hbf, const unsigned short* __restrict__ ebf,
    const int* __restrict__ ei,
    const unsigned short* __restrict__ Wp,      // packed: W1 (8192) | W2 (4096) bf16
    const float* __restrict__ b1, const float* __restrict__ b2,
    float* __restrict__ aggr) {
  __shared__ __align__(16) unsigned short lW[12288];           // 24 KB: W1 | W2
  __shared__ __align__(16) unsigned short lbf[WV][2][16 * UDIM]; // 32 KB restage tiles

  // Cooperative weight stage: 12288 ushorts = 1536 uint4.
  {
    const uint4* s4 = (const uint4*)Wp;
    uint4* d4 = (uint4*)lW;
#pragma unroll
    for (int i = 0; i < 6; i++) d4[threadIdx.x + 256 * i] = s4[threadIdx.x + 256 * i];
  }
  __syncthreads();

  int wave = threadIdx.x >> 5;
  int lane = threadIdx.x & 31;
  int m = lane & 15, hf = lane >> 4;
  int eb = (blockIdx.x * WV + wave) * 32;      // 32 edges per wave, grid exact

  int src0 = ei[eb + m];
  int src1 = ei[eb + 16 + m];
  const unsigned short* hrow0 = hbf + (size_t)src0 * UDIM;
  const unsigned short* hrow1 = hbf + (size_t)src1 * UDIM;
  const unsigned short* erow0 = ebf + (size_t)(eb + m) * UDIM;
  const unsigned short* erow1 = ebf + (size_t)(eb + 16 + m) * UDIM;

  // GEMM1: A = concat[h[src], e] (16x128) for two subtiles, K=128 as 4 slices.
  v8f acc[2][4];
#pragma unroll
  for (int s = 0; s < 2; s++)
#pragma unroll
    for (int t = 0; t < 4; t++) {
      float bias = b1[t * 16 + m];
#pragma unroll
      for (int i = 0; i < 8; i++) acc[s][t][i] = bias;
    }

#pragma unroll
  for (int ks = 0; ks < 4; ks++) {
    v16bf A0 = (ks < 2) ? ldA_bf(hrow0, ks * 32, hf) : ldA_bf(erow0, (ks - 2) * 32, hf);
    v16bf A1 = (ks < 2) ? ldA_bf(hrow1, ks * 32, hf) : ldA_bf(erow1, (ks - 2) * 32, hf);
#pragma unroll
    for (int t = 0; t < 4; t++) {
      v16bf B = ldB_packed(lW, ks, t, lane);
      acc[0][t] = wmma_bf16(A0, B, acc[0][t]);
      acc[1][t] = wmma_bf16(A1, B, acc[1][t]);
    }
  }

  // SiLU -> LDS bf16 tiles (D-layout -> A-layout restage).
#pragma unroll
  for (int s = 0; s < 2; s++)
#pragma unroll
    for (int t = 0; t < 4; t++)
#pragma unroll
      for (int r = 0; r < 8; r++)
        lbf[wave][s][(hf * 8 + r) * UDIM + t * 16 + m] = f2bf_bits(silu_f(acc[s][t][r]));

  // GEMM2: K=64 as 2 slices.
  v8f acc2[2][4];
#pragma unroll
  for (int s = 0; s < 2; s++)
#pragma unroll
    for (int t = 0; t < 4; t++) {
      float bias = b2[t * 16 + m];
#pragma unroll
      for (int i = 0; i < 8; i++) acc2[s][t][i] = bias;
    }

#pragma unroll
  for (int ks = 0; ks < 2; ks++) {
    v16bf Q0 = ldA_bf(&lbf[wave][0][m * UDIM], ks * 32, hf);
    v16bf Q1 = ldA_bf(&lbf[wave][1][m * UDIM], ks * 32, hf);
#pragma unroll
    for (int t = 0; t < 4; t++) {
      v16bf B = ldB_packed(lW + 8192, ks, t, lane);
      acc2[0][t] = wmma_bf16(Q0, B, acc2[0][t]);
      acc2[1][t] = wmma_bf16(Q1, B, acc2[1][t]);
    }
  }

  // Scatter-add (mean division deferred to update kernel).
  int dsts[2][8];
#pragma unroll
  for (int s = 0; s < 2; s++)
#pragma unroll
    for (int r = 0; r < 8; r++)
      dsts[s][r] = ei[N_EDGES + eb + s * 16 + hf * 8 + r];

#pragma unroll
  for (int s = 0; s < 2; s++)
#pragma unroll
    for (int t = 0; t < 4; t++)
#pragma unroll
      for (int r = 0; r < 8; r++)
        atomicAdd(&aggr[(size_t)dsts[s][r] * UDIM + t * 16 + m], acc2[s][t][r]);
}

// ---------------- update kernel: one wave per 16-node tile ----------------
__global__ __launch_bounds__(256) void upd_kernel(
    const unsigned short* __restrict__ hbf, const float* __restrict__ hf32,
    float* __restrict__ aggr, const float* __restrict__ invd,
    const unsigned short* __restrict__ Wp,      // packed: W1 (8192) | W2 (4096) bf16
    const float* __restrict__ b1, const float* __restrict__ b2,
    const float* __restrict__ lng, const float* __restrict__ lnb,
    float* __restrict__ hf32_out, unsigned short* __restrict__ hbf_out) {
  __shared__ __align__(16) unsigned short lW[12288];    // 24 KB
  __shared__ __align__(16) float lfc[WV][16 * UDIM];    // 32 KB; bf16 tile overlaid on front half

  {
    const uint4* s4 = (const uint4*)Wp;
    uint4* d4 = (uint4*)lW;
#pragma unroll
    for (int i = 0; i < 6; i++) d4[threadIdx.x + 256 * i] = s4[threadIdx.x + 256 * i];
  }
  __syncthreads();   // all threads reach this (no early return before it)

  int wave = threadIdx.x >> 5;
  int lane = threadIdx.x & 31;
  int tile = blockIdx.x * WV + wave;
  if (tile < NTILES_N) {                        // wave-uniform guard
    int nb = tile * 16;
    int m = lane & 15, hf = lane >> 4;
    int nodeA = nb + m;

    const unsigned short* hrow = hbf + (size_t)nodeA * UDIM;
    const float* arow = aggr + (size_t)nodeA * UDIM;
    float idg = invd[nodeA];

    v16bf a0 = ldA_bf(hrow, 0, hf);
    v16bf a1 = ldA_bf(hrow, 32, hf);
    v16bf a2 = ldA_f32scaled(arow, 0, hf, idg);
    v16bf a3 = ldA_f32scaled(arow, 32, hf, idg);

    unsigned short* lbfw = (unsigned short*)&lfc[wave][0];   // bf16 intermediate (2 KB of 4 KB)

#pragma unroll
    for (int t = 0; t < 4; t++) {
      float bias = b1[t * 16 + m];
      v8f c;
#pragma unroll
      for (int i = 0; i < 8; i++) c[i] = bias;
      c = wmma_bf16(a0, ldB_packed(lW, 0, t, lane), c);
      c = wmma_bf16(a1, ldB_packed(lW, 1, t, lane), c);
      c = wmma_bf16(a2, ldB_packed(lW, 2, t, lane), c);
      c = wmma_bf16(a3, ldB_packed(lW, 3, t, lane), c);
#pragma unroll
      for (int r = 0; r < 8; r++)
        lbfw[(hf * 8 + r) * UDIM + t * 16 + m] = f2bf_bits(silu_f(c[r]));
    }

    v16bf q0 = ldA_bf(lbfw + m * UDIM, 0, hf);
    v16bf q1 = ldA_bf(lbfw + m * UDIM, 32, hf);

#pragma unroll
    for (int t = 0; t < 4; t++) {
      float bias = b2[t * 16 + m];
      v8f c;
#pragma unroll
      for (int i = 0; i < 8; i++) c[i] = bias;
      c = wmma_bf16(q0, ldB_packed(lW + 8192, 0, t, lane), c);
      c = wmma_bf16(q1, ldB_packed(lW + 8192, 1, t, lane), c);
#pragma unroll
      for (int r = 0; r < 8; r++)
        lfc[wave][(hf * 8 + r) * UDIM + t * 16 + m] = c[r];   // overwrites bf16 region: q0/q1 already read
    }

    // Row phase: residual + LayerNorm, lane = row (lanes 0-15).
    if (lane < 16) {
      int node = nb + m;
      const float* hr = hf32 + (size_t)node * UDIM;
      float* lr = &lfc[wave][m * UDIM];
      float s = 0.0f;
#pragma unroll
      for (int c4 = 0; c4 < UDIM; c4 += 4) {
        float4 hv = *(const float4*)(hr + c4);
        float4 av = *(float4*)(lr + c4);
        float4 vv = make_float4(hv.x + av.x, hv.y + av.y, hv.z + av.z, hv.w + av.w);
        *(float4*)(lr + c4) = vv;
        s += vv.x + vv.y + vv.z + vv.w;
      }
      float mu = s * (1.0f / UDIM);
      float var = 0.0f;
#pragma unroll
      for (int c4 = 0; c4 < UDIM; c4 += 4) {
        float4 vv = *(float4*)(lr + c4);
        float dx = vv.x - mu, dy = vv.y - mu, dz = vv.z - mu, dw = vv.w - mu;
        var += dx * dx + dy * dy + dz * dz + dw * dw;
      }
      var *= (1.0f / UDIM);
      float rs = rsqrtf(var + 1e-5f);
      float* orow = hf32_out + (size_t)node * UDIM;
      unsigned short* brow = hbf_out + (size_t)node * UDIM;
      float* az = aggr + (size_t)node * UDIM;
#pragma unroll
      for (int c = 0; c < UDIM; c++) {
        float v = (lr[c] - mu) * rs * lng[c] + lnb[c];
        orow[c] = v;
        brow[c] = f2bf_bits(v);
      }
      float4 z = make_float4(0.f, 0.f, 0.f, 0.f);
#pragma unroll
      for (int c4 = 0; c4 < UDIM; c4 += 4) *(float4*)(az + c4) = z;   // reset aggr for next layer
    }
  }
}

// ---------------- host launcher ----------------
extern "C" void kernel_launch(void* const* d_in, const int* in_sizes, int n_in,
                              void* d_out, int out_size, void* d_ws, size_t ws_size,
                              hipStream_t stream) {
  (void)in_sizes; (void)n_in; (void)out_size; (void)ws_size;
  const float* x     = (const float*)d_in[0];
  const int*   ei    = (const int*)d_in[1];
  const float* eattr = (const float*)d_in[2];
  const float* nodeW = (const float*)d_in[3];
  const float* nodeB = (const float*)d_in[4];
  const float* edgeW = (const float*)d_in[5];
  const float* edgeB = (const float*)d_in[6];
  const float* msgW1 = (const float*)d_in[7];
  const float* msgB1 = (const float*)d_in[8];
  const float* msgW2 = (const float*)d_in[9];
  const float* msgB2 = (const float*)d_in[10];
  const float* updW1 = (const float*)d_in[11];
  const float* updB1 = (const float*)d_in[12];
  const float* updW2 = (const float*)d_in[13];
  const float* updB2 = (const float*)d_in[14];
  const float* lng   = (const float*)d_in[15];
  const float* lnb   = (const float*)d_in[16];
  float* out = (float*)d_out;

  char* ws = (char*)d_ws;
  size_t off = 0;
  auto carve = [&](size_t bytes) -> char* {
    char* p = ws + off;
    off = (off + bytes + 255) & ~(size_t)255;
    return p;
  };
  float*          h_f32 = (float*)carve((size_t)N_NODES * UDIM * 4);
  unsigned short* h_bf  = (unsigned short*)carve((size_t)N_NODES * UDIM * 2);
  unsigned short* e_bf  = (unsigned short*)carve((size_t)N_EDGES * UDIM * 2);
  float*          aggr  = (float*)carve((size_t)N_NODES * UDIM * 4);
  float*          deg   = (float*)carve((size_t)N_NODES * 4);
  float*          invd  = (float*)carve((size_t)N_NODES * 4);
  unsigned short* pW    = (unsigned short*)carve((size_t)DEPTH * 24576 * 2);

  // Pack weights into WMMA B-operand layout (per layer: msgW1 | msgW2 | updW1 | updW2).
  for (int L = 0; L < DEPTH; L++) {
    unsigned short* base = pW + (size_t)L * 24576;
    pack_w_kernel<<<32, 256, 0, stream>>>(msgW1 + (size_t)L * 8192, base,          128);
    pack_w_kernel<<<16, 256, 0, stream>>>(msgW2 + (size_t)L * 4096, base + 8192,    64);
    pack_w_kernel<<<32, 256, 0, stream>>>(updW1 + (size_t)L * 8192, base + 12288,  128);
    pack_w_kernel<<<16, 256, 0, stream>>>(updW2 + (size_t)L * 4096, base + 20480,   64);
  }

  zero_kernel<<<((size_t)N_NODES * 65 + 255) / 256, 256, 0, stream>>>(aggr, deg);
  node_enc_kernel<<<(N_NODES + 255) / 256, 256, 0, stream>>>(x, nodeW, nodeB, h_f32, h_bf);
  edge_enc_kernel<<<(N_EDGES + 255) / 256, 256, 0, stream>>>(eattr, ei, edgeW, edgeB, e_bf, deg);
  invdeg_kernel<<<(N_NODES + 255) / 256, 256, 0, stream>>>(deg, invd);

  for (int L = 0; L < DEPTH; L++) {
    unsigned short* base = pW + (size_t)L * 24576;
    // 8 waves/block * 2 tiles/wave * 16 edges = 256 edges per block -> 3125 blocks exact.
    msg_kernel<<<NTILES_E / (WV * 2), 256, 0, stream>>>(h_bf, e_bf, ei, base,
                                                        msgB1 + L * 64, msgB2 + L * 64, aggr);
    float* hout = (L == DEPTH - 1) ? out : h_f32;
    upd_kernel<<<(NTILES_N + WV - 1) / WV, 256, 0, stream>>>(
        h_bf, h_f32, aggr, invd, base + 12288,
        updB1 + L * 64, updB2 + L * 64, lng + L * 64, lnb + L * 64, hout, h_bf);
  }
}
